// VoxelBackBoneDeconvRes_30322469109766
// MI455X (gfx1250) — matrix operand order repaired and checked by reading, not compiled, not run on hardware
//
#include <hip/hip_runtime.h>
#include <hip/hip_bf16.h>
#include <stdint.h>

// ---------------------------------------------------------------------------
// Types for CDNA5 WMMA (wave32, v_wmma_f32_16x16x32_bf16)
// ---------------------------------------------------------------------------
typedef __attribute__((ext_vector_type(16))) __bf16         v16bf;
typedef __attribute__((ext_vector_type(16))) unsigned short v16us;
typedef __attribute__((ext_vector_type(8)))  unsigned short v8us;
typedef __attribute__((ext_vector_type(8)))  float          v8f;
typedef __attribute__((ext_vector_type(4)))  float          f4;

__device__ __forceinline__ unsigned short f32_to_bf16(float f) {
  unsigned int u = __builtin_bit_cast(unsigned int, f);
  u += 0x7FFFu + ((u >> 16) & 1u);            // round-to-nearest-even
  return (unsigned short)(u >> 16);
}

// Coordinate mapping: output coord + tap -> input coord
//   MODE 0: stride-1 conv, pad 1
//   MODE 1: stride-2 conv, pad 1
//   MODE 2: transposed conv (lhs_dilation 2, pad 1)
template <int MODE>
__device__ __forceinline__ bool map_coord(int o, int k, int In, int& i) {
  if (MODE == 0) { i = o + k - 1;     return (i >= 0) && (i < In); }
  if (MODE == 1) { i = 2 * o + k - 1; return (i >= 0) && (i < In); }
  int d = o + k - 1;                  // position in dilated input
  i = d >> 1;
  return (d >= 0) && ((d & 1) == 0) && (i < In);
}

// ---------------------------------------------------------------------------
// Utility kernels
// ---------------------------------------------------------------------------
__global__ void __launch_bounds__(256) zero_f32(float* __restrict__ p, long n) {
  long i = (long)blockIdx.x * blockDim.x + threadIdx.x;
  long s = (long)gridDim.x * blockDim.x;
  for (; i < n; i += s) p[i] = 0.f;
}

__global__ void __launch_bounds__(256)
scatter_voxels(const float* __restrict__ feat, const int* __restrict__ coords,
               float* __restrict__ x0, float* __restrict__ m0, int n, long plane) {
  int i = blockIdx.x * blockDim.x + threadIdx.x;
  if (i >= n) return;
  int z = coords[4 * i + 1], y = coords[4 * i + 2], x = coords[4 * i + 3];
  long s = ((long)z * 193 + y) * 193 + x;
  m0[s] = 1.f;
#pragma unroll
  for (int c = 0; c < 4; ++c) x0[(long)c * plane + s] = feat[4 * i + c];
}

// mask dilation: mo = (conv3x3x3(mi, ones) > 0)
template <int MODE>
__global__ void __launch_bounds__(256)
mask_conv(const float* __restrict__ mi, float* __restrict__ mo,
          int Di, int Hi, int Wi, int Do, int Ho, int Wo) {
  long idx = (long)blockIdx.x * blockDim.x + threadIdx.x;
  long nv = (long)Do * Ho * Wo;
  if (idx >= nv) return;
  int x = (int)(idx % Wo);
  int y = (int)((idx / Wo) % Ho);
  int z = (int)(idx / ((long)Wo * Ho));
  float any = 0.f;
  for (int kz = 0; kz < 3; ++kz) {
    int iz; if (!map_coord<MODE>(z, kz, Di, iz)) continue;
    for (int ky = 0; ky < 3; ++ky) {
      int iy; if (!map_coord<MODE>(y, ky, Hi, iy)) continue;
      for (int kx = 0; kx < 3; ++kx) {
        int ix; if (!map_coord<MODE>(x, kx, Wi, ix)) continue;
        if (mi[((long)iz * Hi + iy) * Wi + ix] > 0.f) any = 1.f;
      }
    }
  }
  mo[idx] = any;
}

// pack OIDHW f32 weights -> [tap][Cout][Cpad] bf16, zero-padded channels
__global__ void __launch_bounds__(256)
pack_weights(const float* __restrict__ w, unsigned short* __restrict__ dst,
             int Cout, int Cin, int Cpad) {
  int j = blockIdx.x * blockDim.x + threadIdx.x;
  int total = 27 * Cout * Cpad;
  if (j >= total) return;
  int c   = j % Cpad;
  int oc  = (j / Cpad) % Cout;
  int tap = j / (Cpad * Cout);
  float v = (c < Cin) ? w[((long)oc * Cin + c) * 27 + tap] : 0.f;
  dst[j] = f32_to_bf16(v);
}

// ---------------------------------------------------------------------------
// LDS-staged implicit-GEMM 3x3x3 conv, bf16 WMMA, fused BN+ReLU+mask.
//   Block = 128 threads (4 waves), one (zo,yo) output row segment of XT x.
//   Stage: 9 halo rows x ROWLEN x CPAD bf16 into LDS once; all 27 taps and
//   all Cout tiles reuse it. A fragment = 2 x ds_load_b128 per lane;
//   B fragment = 2 x global_load_b128 from the pre-packed bf16 weights;
//   then one v_wmma_f32_16x16x32_bf16 per K-chunk.
//   Pad channels (c >= CIN) stage clamped-but-finite data; the packed B is
//   zero there, so their WMMA contribution is exactly zero.
// ---------------------------------------------------------------------------
template <int MODE, int CIN>
__global__ void __launch_bounds__(128)
conv3d_wmma(const float* __restrict__ in, const unsigned short* __restrict__ wp,
            const float* __restrict__ gamma, const float* __restrict__ beta,
            const float* __restrict__ maskO, const float* __restrict__ mask2,
            float* __restrict__ out, int Cout,
            int Di, int Hi, int Wi, int Do, int Ho, int Wo) {
  constexpr int CPAD   = ((CIN + 31) / 32) * 32;       // K padded to chunk size
  constexpr int XT     = (MODE == 1) ? 32 : 64;        // output x per block
  constexpr int ROWLEN = (MODE == 2) ? 34 : 66;        // staged input row length
  constexpr int KCH    = CPAD / 32;                    // K-chunks per tap
  constexpr int CP     = CPAD + 8;                     // padded LDS row stride
  __shared__ __align__(16) unsigned short smem[9 * ROWLEN * CP];

  const int zo = blockIdx.z, yo = blockIdx.y;
  const int xb = blockIdx.x * XT;
  const long ivol = (long)Di * Hi * Wi;
  const int bix = (MODE == 0) ? (xb - 1) : (MODE == 1) ? (2 * xb - 1) : ((xb - 1) >> 1);

  // ---- cooperative staging: clamp+select (no predicated loads), f32->bf16 ----
  for (int j = threadIdx.x; j < 9 * ROWLEN * (CPAD / 4); j += 128) {
    int c0 = (j % (CPAD / 4)) * 4;
    int t  = (j / (CPAD / 4)) % ROWLEN;
    int r  = j / ((CPAD / 4) * ROWLEN);
    int kz = r / 3, ky = r - kz * 3;
    int iz, iy;
    bool vz = map_coord<MODE>(zo, kz, Di, iz);
    bool vy = map_coord<MODE>(yo, ky, Hi, iy);
    int ix = bix + t;
    float f = (vz && vy && (ix >= 0) && (ix < Wi)) ? 1.f : 0.f;
    int izc = min(max(iz, 0), Di - 1);
    int iyc = min(max(iy, 0), Hi - 1);
    int ixc = min(max(ix, 0), Wi - 1);
    const float* src = in + (((long)izc * Hi + iyc) * Wi + ixc);
    unsigned int pk[2];
#pragma unroll
    for (int u = 0; u < 2; ++u) {
      int ca = c0 + 2 * u, cb = ca + 1;
      // compile-time clamp: pad channels re-read channel CIN-1 (finite; B=0 there)
      unsigned short qa = f32_to_bf16(src[(long)(ca < CIN ? ca : CIN - 1) * ivol] * f);
      unsigned short qb = f32_to_bf16(src[(long)(cb < CIN ? cb : CIN - 1) * ivol] * f);
      pk[u] = (unsigned int)qa | ((unsigned int)qb << 16);
    }
    *(uint2*)&smem[(r * ROWLEN + t) * CP + c0] = make_uint2(pk[0], pk[1]);
  }
  __syncthreads();

  // ---- compute phase ----
  const int lane = threadIdx.x & 31;
  const int wave = threadIdx.x >> 5;
  const int hi   = lane >> 4;            // lane half (K sub-range per ISA layout)
  const int ml   = lane & 15;            // A row / D column within tile
  const int wsub  = (MODE == 1) ? (wave & 1) : wave;   // x-subtile of 16
  const int ctst  = (MODE == 1) ? (wave >> 1) : 0;     // ct split across waves
  const int ctinc = (MODE == 1) ? 2 : 1;
  const int lx = wsub * 16 + ml;         // local output x
  const int CT = Cout >> 4;

  const long ovol = (long)Do * Ho * Wo;
  const long rowb = ((long)zo * Ho + yo) * Wo;
  const v8us vzero = {};

  for (int ct = ctst; ct < CT; ct += ctinc) {
    v8f acc = {0.f, 0.f, 0.f, 0.f, 0.f, 0.f, 0.f, 0.f};
    const int oc = ct * 16 + ml;

    for (int r = 0; r < 9; ++r) {
#pragma unroll
      for (int kx = 0; kx < 3; ++kx) {
        int t; bool pok = true;
        if (MODE == 0)      t = lx + kx;
        else if (MODE == 1) t = 2 * lx + kx;
        else { int d = xb + lx + kx - 1; pok = (d >= 0) && ((d & 1) == 0); t = (d >> 1) - bix; }
        const int tap = r * 3 + kx;
        const unsigned short* arow = &smem[(r * ROWLEN + t) * CP];
        const unsigned short* brow = &wp[((long)tap * Cout + oc) * CPAD];
#pragma unroll
        for (int kc = 0; kc < KCH; ++kc) {
          // A (16x32 bf16): lane half h holds K = 8h+{0..7} and 16+8h+{0..7}
          const v8us* ap = (const v8us*)(arow + kc * 32 + hi * 8);
          v8us a0 = ap[0];
          v8us a1 = ap[2];
          if constexpr (MODE == 2) { a0 = pok ? a0 : vzero; a1 = pok ? a1 : vzero; }
          // B (32x16 bf16): lane half h holds K = 8h+{0..7} and 16+8h+{0..7}
          const v8us* bp = (const v8us*)(brow + kc * 32 + hi * 8);
          v8us b0 = bp[0], b1 = bp[2];
          v16us au = __builtin_shufflevector(a0, a1, 0,1,2,3,4,5,6,7,8,9,10,11,12,13,14,15);
          v16us bu = __builtin_shufflevector(b0, b1, 0,1,2,3,4,5,6,7,8,9,10,11,12,13,14,15);
          acc = __builtin_amdgcn_wmma_f32_16x16x32_bf16(
              false, __builtin_bit_cast(v16bf, au),
              false, __builtin_bit_cast(v16bf, bu),
              (short)0, acc, false, false);
        }
      }
    }

    // fused BN + ReLU + mask epilogue. D: VGPR q <-> M = q + 8*hi, N = ml.
    // Each lane's 8 outputs are consecutive in x -> vector loads/stores on the
    // interior fast path (wave-uniform branch; no per-lane predication).
    const float g = gamma[oc], bb = beta[oc];
    if (xb + XT <= Wo) {
      const long spb = rowb + xb + wsub * 16 + hi * 8;
      f4 m0v, m1v;
      __builtin_memcpy(&m0v, &maskO[spb], 16);
      __builtin_memcpy(&m1v, &maskO[spb + 4], 16);
      if (mask2) {
        f4 n0, n1;
        __builtin_memcpy(&n0, &mask2[spb], 16);
        __builtin_memcpy(&n1, &mask2[spb + 4], 16);
        m0v *= n0; m1v *= n1;
      }
      f4 r0, r1;
#pragma unroll
      for (int q = 0; q < 4; ++q) {
        r0[q] = fmaxf(acc[q] * g + bb, 0.f) * m0v[q];
        r1[q] = fmaxf(acc[q + 4] * g + bb, 0.f) * m1v[q];
      }
      float* op = out + (long)oc * ovol + spb;
      __builtin_memcpy(op, &r0, 16);
      __builtin_memcpy(op + 4, &r1, 16);
    } else {
#pragma unroll
      for (int q = 0; q < 8; ++q) {
        int xo_ = xb + wsub * 16 + q + hi * 8;
        int xcl = min(xo_, Wo - 1);
        long sp = rowb + xcl;
        float mv = maskO[sp];
        if (mask2) mv *= mask2[sp];
        float v = fmaxf(acc[q] * g + bb, 0.f) * mv;
        if (xo_ < Wo) out[(long)oc * ovol + sp] = v;
      }
    }
  }
}

// ---------------------------------------------------------------------------
// Host-side orchestration
// ---------------------------------------------------------------------------
namespace {

constexpr long P0 = 21L * 193 * 193;   // full-res voxels  (782229)
constexpr long P1 = 11L * 97 * 97;     // stride-2         (103499)
constexpr long P2 = 6L * 49 * 49;      // stride-4         (14406)

// workspace bump offsets (floats)
constexpr long o_x0   = 0;                      // 4*P0
constexpr long o_m0   = o_x0  + 4 * P0;         // P0
constexpr long o_m1   = o_m0  + P0;             // P0
constexpr long o_x1   = o_m1  + P0;             // 16*P0
constexpr long o_m11  = o_x1  + 16 * P0;        // P0
constexpr long o_md1  = o_m11 + P0;             // P0
constexpr long o_cat1 = o_md1 + P0;             // 48*P0  [xd1 | xe1*md1]
constexpr long o_m2   = o_cat1 + 48 * P0;       // P1
constexpr long o_x2   = o_m2  + P1;             // 32*P1
constexpr long o_x2b  = o_x2  + 32 * P1;        // 32*P1
constexpr long o_m22  = o_x2b + 32 * P1;        // P1
constexpr long o_md2  = o_m22 + P1;             // P1
constexpr long o_cat2 = o_md2 + P1;             // 64*P1  [xd2 | xe2*md2]
constexpr long o_x2d  = o_cat2 + 64 * P1;       // 32*P1
constexpr long o_m3   = o_x2d + 32 * P1;        // P2
constexpr long o_x3   = o_m3  + P2;             // 64*P2
constexpr long o_x3b  = o_x3  + 64 * P2;        // 64*P2
constexpr long o_wpk  = o_x3b + 64 * P2;        // packed bf16 weights (~0.9 MB)

struct Dim3i { int d, h, w; };
constexpr Dim3i FULL{21, 193, 193};
constexpr Dim3i HALF{11, 97, 97};
constexpr Dim3i QUAR{6, 49, 49};

void run_mask(int mode, const float* mi, float* mo, Dim3i di, Dim3i dout, hipStream_t s) {
  long n = (long)dout.d * dout.h * dout.w;
  int blocks = (int)((n + 255) / 256);
  if (mode == 0)
    mask_conv<0><<<blocks, 256, 0, s>>>(mi, mo, di.d, di.h, di.w, dout.d, dout.h, dout.w);
  else if (mode == 1)
    mask_conv<1><<<blocks, 256, 0, s>>>(mi, mo, di.d, di.h, di.w, dout.d, dout.h, dout.w);
  else
    mask_conv<2><<<blocks, 256, 0, s>>>(mi, mo, di.d, di.h, di.w, dout.d, dout.h, dout.w);
}

void run_conv(int mode, int cin, const float* in, const unsigned short* wp,
              const float* g, const float* b, const float* mask, const float* mask2,
              float* out, int Cout, Dim3i di, Dim3i dout, hipStream_t s) {
  int xt = (mode == 1) ? 32 : 64;
  dim3 grid((dout.w + xt - 1) / xt, dout.h, dout.d), blk(128);
#define LAUNCH(M, C)                                                            \
  conv3d_wmma<M, C><<<grid, blk, 0, s>>>(in, wp, g, b, mask, mask2, out, Cout,  \
                                         di.d, di.h, di.w, dout.d, dout.h, dout.w)
  if (mode == 0) {
    if      (cin == 4)  LAUNCH(0, 4);
    else if (cin == 16) LAUNCH(0, 16);
    else if (cin == 32) LAUNCH(0, 32);
    else if (cin == 48) LAUNCH(0, 48);
    else                LAUNCH(0, 64);
  } else if (mode == 1) {
    if (cin == 16) LAUNCH(1, 16);
    else           LAUNCH(1, 32);
  } else {
    if (cin == 32) LAUNCH(2, 32);
    else           LAUNCH(2, 64);
  }
#undef LAUNCH
}

} // namespace

extern "C" void kernel_launch(void* const* d_in, const int* in_sizes, int n_in,
                              void* d_out, int out_size, void* d_ws, size_t ws_size,
                              hipStream_t stream) {
  (void)in_sizes; (void)n_in; (void)out_size; (void)ws_size;
  const float* vf = (const float*)d_in[0];
  const int*   vc = (const int*)d_in[1];
  // params flattened in dict insertion order: (w,g,b) x {1,2a,2b,3a,3b,d2,22,c2,d1,11,c1}
  const float *W1  = (const float*)d_in[2],  *G1  = (const float*)d_in[3],  *B1  = (const float*)d_in[4];
  const float *W2a = (const float*)d_in[5],  *G2a = (const float*)d_in[6],  *B2a = (const float*)d_in[7];
  const float *W2b = (const float*)d_in[8],  *G2b = (const float*)d_in[9],  *B2b = (const float*)d_in[10];
  const float *W3a = (const float*)d_in[11], *G3a = (const float*)d_in[12], *B3a = (const float*)d_in[13];
  const float *W3b = (const float*)d_in[14], *G3b = (const float*)d_in[15], *B3b = (const float*)d_in[16];
  const float *Wd2 = (const float*)d_in[17], *Gd2 = (const float*)d_in[18], *Bd2 = (const float*)d_in[19];
  const float *W22 = (const float*)d_in[20], *G22 = (const float*)d_in[21], *B22 = (const float*)d_in[22];
  const float *Wc2 = (const float*)d_in[23], *Gc2 = (const float*)d_in[24], *Bc2 = (const float*)d_in[25];
  const float *Wd1 = (const float*)d_in[26], *Gd1 = (const float*)d_in[27], *Bd1 = (const float*)d_in[28];
  const float *W11 = (const float*)d_in[29], *G11 = (const float*)d_in[30], *B11 = (const float*)d_in[31];
  const float *Wc1 = (const float*)d_in[32], *Gc1 = (const float*)d_in[33], *Bc1 = (const float*)d_in[34];

  float* ws  = (float*)d_ws;
  float* out = (float*)d_out;

  // ---- pack all weights to bf16 [tap][Cout][Cpad] ----
  unsigned short* wpk = (unsigned short*)(ws + o_wpk);
  long woff = 0;
  auto pack = [&](const float* w, int cout, int cin, int cpad) -> unsigned short* {
    unsigned short* dst = wpk + woff;
    int n = 27 * cout * cpad;
    pack_weights<<<(n + 255) / 256, 256, 0, stream>>>(w, dst, cout, cin, cpad);
    woff += n;
    return dst;
  };
  unsigned short* P_w1  = pack(W1,  16, 4,  32);
  unsigned short* P_w2a = pack(W2a, 32, 16, 32);
  unsigned short* P_w2b = pack(W2b, 32, 32, 32);
  unsigned short* P_w3a = pack(W3a, 64, 32, 32);
  unsigned short* P_w3b = pack(W3b, 64, 64, 64);
  unsigned short* P_wd2 = pack(Wd2, 32, 64, 64);
  unsigned short* P_w22 = pack(W22, 32, 32, 32);
  unsigned short* P_wc2 = pack(Wc2, 32, 64, 64);
  unsigned short* P_wd1 = pack(Wd1, 32, 32, 32);
  unsigned short* P_w11 = pack(W11, 16, 16, 32);
  unsigned short* P_wc1 = pack(Wc1, 32, 48, 64);

  // ---- zero scatter targets (x0, m0 contiguous: 5*P0 floats) ----
  {
    long n = 5 * P0;
    zero_f32<<<(int)((n + 255) / 256), 256, 0, stream>>>(ws + o_x0, n);
  }
  // ---- scatter voxels into dense grid + occupancy mask ----
  scatter_voxels<<<(80000 + 255) / 256, 256, 0, stream>>>(vf, vc, ws + o_x0, ws + o_m0, 80000, P0);

  // stage 1 (full res): m1 = dilate(m0); x1 = conv(x0, w1) 4->16
  run_mask(0, ws + o_m0, ws + o_m1, FULL, FULL, stream);
  run_conv(0, 4, ws + o_x0, P_w1, G1, B1, ws + o_m1, nullptr, ws + o_x1, 16, FULL, FULL, stream);

  // stage 2 (stride 2 -> half): m2; x2 = conv_s2(x1, w2a) 16->32; x2b = subm(x2, w2b)
  run_mask(1, ws + o_m1, ws + o_m2, FULL, HALF, stream);
  run_conv(1, 16, ws + o_x1, P_w2a, G2a, B2a, ws + o_m2, nullptr, ws + o_x2, 32, FULL, HALF, stream);
  run_conv(0, 32, ws + o_x2, P_w2b, G2b, B2b, ws + o_m2, nullptr, ws + o_x2b, 32, HALF, HALF, stream);

  // stage 3 (stride 2 -> quarter): m3; x3 = conv_s2(x2b, w3a) 32->64; x3b = subm(x3, w3b)
  run_mask(1, ws + o_m2, ws + o_m3, HALF, QUAR, stream);
  run_conv(1, 32, ws + o_x2b, P_w3a, G3a, B3a, ws + o_m3, nullptr, ws + o_x3, 64, HALF, QUAR, stream);
  run_conv(0, 64, ws + o_x3, P_w3b, G3b, B3b, ws + o_m3, nullptr, ws + o_x3b, 64, QUAR, QUAR, stream);

  // decoder level 2: md2; cat2 = [deconv(x3b, wd2) | spconv1(x2b, w22)*md2]; x2d = subm(cat2, wc2)
  run_mask(2, ws + o_m3, ws + o_md2, QUAR, HALF, stream);
  run_conv(2, 64, ws + o_x3b, P_wd2, Gd2, Bd2, ws + o_md2, nullptr, ws + o_cat2, 32, QUAR, HALF, stream);
  run_mask(0, ws + o_m2, ws + o_m22, HALF, HALF, stream);
  run_conv(0, 32, ws + o_x2b, P_w22, G22, B22, ws + o_m22, ws + o_md2,
           ws + o_cat2 + 32 * P1, 32, HALF, HALF, stream);
  run_conv(0, 64, ws + o_cat2, P_wc2, Gc2, Bc2, ws + o_md2, nullptr, ws + o_x2d, 32, HALF, HALF, stream);

  // decoder level 1: md1; cat1 = [deconv(x2d, wd1) | spconv1(x1, w11)*md1]
  run_mask(2, ws + o_md2, ws + o_md1, HALF, FULL, stream);
  run_conv(2, 32, ws + o_x2d, P_wd1, Gd1, Bd1, ws + o_md1, nullptr, ws + o_cat1, 32, HALF, FULL, stream);
  run_mask(0, ws + o_m1, ws + o_m11, FULL, FULL, stream);
  run_conv(0, 16, ws + o_x1, P_w11, G11, B11, ws + o_m11, ws + o_md1,
           ws + o_cat1 + 32 * P0, 16, FULL, FULL, stream);

  // output: out = subm(cat1, wc1) 48->32, mask md1 -> d_out (32*P0 floats, NCDHW)
  run_conv(0, 48, ws + o_cat1, P_wc1, Gc1, Bc1, ws + o_md1, nullptr, out, 32, FULL, FULL, stream);
}
